// GridEncoder_14422500180315
// MI455X (gfx1250) — matrix operand order repaired and checked by reading, not compiled, not run on hardware
//
#include <hip/hip_runtime.h>
#include <hip/hip_fp16.h>
#include <hip/hip_bf16.h>

typedef _Float16 half_t;
typedef __attribute__((ext_vector_type(16))) _Float16 v16h;
typedef __attribute__((ext_vector_type(8)))  _Float16 v8h;
typedef __attribute__((ext_vector_type(8)))  float    v8f;

#define B_   64
#define G_   128
#define P_   8
#define NC_  256
#define D_   256
#define L_   6
#define NH_  8
#define HD_  32
#define MLP_ 1024
#define NPH_ 16
#define N_   256
#define MT_  (B_*N_)   // 16384 rows of residual stream

// ---------------------------------------------------------------------------
// Weight prep: f32 W (K x N) -> f16 W^T (N x K), so each output column's K
// values are contiguous (B-fragment per lane = one 32B load).
// ---------------------------------------------------------------------------
__global__ void trans_k(const float* __restrict__ W, half_t* __restrict__ WT,
                        int K, int Nn) {
  size_t idx = (size_t)blockIdx.x * blockDim.x + threadIdx.x;
  if (idx >= (size_t)K * Nn) return;
  int n = (int)(idx % Nn);
  int k = (int)(idx / Nn);
  WT[(size_t)n * K + k] = (half_t)W[idx];
}

// ---------------------------------------------------------------------------
// Patch embed gather: grid indices -> color embeddings, patchified to
// (B*N, 1024) f16 row-major (A operand of the projection GEMM).
// ---------------------------------------------------------------------------
__global__ void embed_k(const int* __restrict__ grd, const float* __restrict__ ce,
                        half_t* __restrict__ e) {
  size_t idx = (size_t)blockIdx.x * blockDim.x + threadIdx.x;
  if (idx >= (size_t)MT_ * 1024) return;
  int f   = (int)(idx & 1023);
  int tok = (int)((idx >> 10) & 255);
  int b   = (int)(idx >> 18);
  int c  = f & 15;
  int ix = (f >> 4) & 7;
  int iy = f >> 7;
  int px = tok & 15, py = tok >> 4;
  int gy = py * P_ + iy, gx = px * P_ + ix;
  int col = grd[((size_t)b * G_ + gy) * G_ + gx];
  e[idx] = (half_t)ce[col * 16 + c];
}

// ---------------------------------------------------------------------------
// LayerNorm: wave per row (D=256 -> 8 floats/lane), shfl_xor reduction.
// Writes f16 (GEMM A operand) or f32 (final output).
// ---------------------------------------------------------------------------
template<bool OUTF32>
__global__ __launch_bounds__(256)
void ln_k(const float* __restrict__ x, const float* __restrict__ g,
          const float* __restrict__ be, void* __restrict__ out, int Mtot) {
  const int lane = threadIdx.x & 31;
  const int wave = threadIdx.x >> 5;
  const int row  = blockIdx.x * 8 + wave;
  if (row >= Mtot) return;
  const float* xr = x + (size_t)row * D_;
  float v[8];
  float s = 0.f;
#pragma unroll
  for (int i = 0; i < 8; ++i) { v[i] = xr[lane * 8 + i]; s += v[i]; }
#pragma unroll
  for (int m = 16; m >= 1; m >>= 1) s += __shfl_xor(s, m, 32);
  const float mean = s * (1.0f / D_);
  float t = 0.f;
#pragma unroll
  for (int i = 0; i < 8; ++i) { float d = v[i] - mean; t += d * d; }
#pragma unroll
  for (int m = 16; m >= 1; m >>= 1) t += __shfl_xor(t, m, 32);
  const float rs = rsqrtf(t * (1.0f / D_) + 1e-5f);
#pragma unroll
  for (int i = 0; i < 8; ++i) {
    const int colj = lane * 8 + i;
    const float val = (v[i] - mean) * rs * g[colj] + be[colj];
    if (OUTF32) ((float*)out)[(size_t)row * D_ + colj] = val;
    else        ((half_t*)out)[(size_t)row * D_ + colj] = (half_t)val;
  }
}

// ---------------------------------------------------------------------------
// Generic WMMA GEMM: C = act(A(Mtot x K, f16 rm) @ WT^T + bias [+ resid])
// One wave computes a 32x64 tile: per k-step, 2 A-fragments + 4 B-fragments
// feed 8 v_wmma_f32_16x16x32_f16 (B reused across both M halves -> ~42
// FLOP/byte from L2, vs 26 for a 16x64 tile). Weight stream is prefetched
// one K-block ahead (global_prefetch_b8).
// Fragment layouts (A and B both use contiguous k; C as in CDNA5 ISA 7.12.2):
//  A lane L: row = L&15; elems 0..7 -> K = kb + (L>=16?16:0) + 0..7
//            elems 8..15 -> K = kb + (L>=16?16:0) + 8..15
//  B lane L: col = L&15; elems i -> K = kb + (L>=16?16:0) + i (contiguous)
//  C lane L: col = L&15; vgpr r -> row = r + (L>=16?8:0)
// OMODE: 0=f32 rowmajor, 1=f16 rowmajor, 2=f16 (b,h,n,d), 3=f16 (b,h,d,n)
// ---------------------------------------------------------------------------
template<int OMODE, bool GELU, bool RESID>
__global__ __launch_bounds__(256)
void gemm_f16(const half_t* __restrict__ A, const half_t* __restrict__ WT,
              const float* __restrict__ bias, const float* __restrict__ resid,
              void* __restrict__ outp, int Mtot, int K, int Nout) {
  const int lane = threadIdx.x & 31;
  const int wave = threadIdx.x >> 5;
  const int tiles_n = Nout >> 6;
  const int total   = (Mtot >> 5) * tiles_n;
  const int tile = blockIdx.x * 8 + wave;
  if (tile >= total) return;
  const int tm  = tile / tiles_n;   // 32-row block
  const int tn  = tile % tiles_n;   // 64-col block
  const int hi  = lane >> 4;
  const int l15 = lane & 15;

  const half_t* arow0 = A + (size_t)(tm * 32 + l15) * K + hi * 16;
  const half_t* arow1 = arow0 + (size_t)16 * K;
  const half_t* b0 = WT + (size_t)(tn * 64 +  0 + l15) * K + hi * 16;
  const half_t* b1 = WT + (size_t)(tn * 64 + 16 + l15) * K + hi * 16;
  const half_t* b2 = WT + (size_t)(tn * 64 + 32 + l15) * K + hi * 16;
  const half_t* b3 = WT + (size_t)(tn * 64 + 48 + l15) * K + hi * 16;

  v8f acc[8] = {};
  for (int kb = 0; kb < K; kb += 32) {
    union { v16h v; v8h h2[2]; } ua0, ua1;
    ua0.h2[0] = *(const v8h*)(arow0 + kb);
    ua0.h2[1] = *(const v8h*)(arow0 + kb + 8);
    ua1.h2[0] = *(const v8h*)(arow1 + kb);
    ua1.h2[1] = *(const v8h*)(arow1 + kb + 8);
    v16h bv0 = *(const v16h*)(b0 + kb);
    v16h bv1 = *(const v16h*)(b1 + kb);
    v16h bv2 = *(const v16h*)(b2 + kb);
    v16h bv3 = *(const v16h*)(b3 + kb);
    if (kb + 32 < K) {   // pull next K-block of the weight stream toward L2/L0
      __builtin_prefetch(b0 + kb + 32, 0, 1);
      __builtin_prefetch(b2 + kb + 32, 0, 1);
    }
    acc[0] = __builtin_amdgcn_wmma_f32_16x16x32_f16(false, ua0.v, false, bv0, (short)0, acc[0], false, false);
    acc[1] = __builtin_amdgcn_wmma_f32_16x16x32_f16(false, ua1.v, false, bv0, (short)0, acc[1], false, false);
    acc[2] = __builtin_amdgcn_wmma_f32_16x16x32_f16(false, ua0.v, false, bv1, (short)0, acc[2], false, false);
    acc[3] = __builtin_amdgcn_wmma_f32_16x16x32_f16(false, ua1.v, false, bv1, (short)0, acc[3], false, false);
    acc[4] = __builtin_amdgcn_wmma_f32_16x16x32_f16(false, ua0.v, false, bv2, (short)0, acc[4], false, false);
    acc[5] = __builtin_amdgcn_wmma_f32_16x16x32_f16(false, ua1.v, false, bv2, (short)0, acc[5], false, false);
    acc[6] = __builtin_amdgcn_wmma_f32_16x16x32_f16(false, ua0.v, false, bv3, (short)0, acc[6], false, false);
    acc[7] = __builtin_amdgcn_wmma_f32_16x16x32_f16(false, ua1.v, false, bv3, (short)0, acc[7], false, false);
  }

#pragma unroll
  for (int j = 0; j < 4; ++j) {
    const int ncol = tn * 64 + j * 16 + l15;
    const float bb = bias[ncol];
#pragma unroll
    for (int mh = 0; mh < 2; ++mh) {
      const v8f c = acc[j * 2 + mh];
#pragma unroll
      for (int r = 0; r < 8; ++r) {
        const int m = tm * 32 + mh * 16 + hi * 8 + r;
        float val = c[r] + bb;
        if (GELU)  val = 0.5f * val * (1.0f + erff(val * 0.70710678118f));
        if (RESID) val += resid[(size_t)m * Nout + ncol];
        if (OMODE == 0) {
          ((float*)outp)[(size_t)m * Nout + ncol] = val;
        } else if (OMODE == 1) {
          ((half_t*)outp)[(size_t)m * Nout + ncol] = (half_t)val;
        } else {
          const int bidx = m >> 8, n = m & 255;     // N_ == 256
          const int hh = ncol >> 5, d = ncol & 31;  // HD_ == 32
          if (OMODE == 2)
            ((half_t*)outp)[(((size_t)(bidx * NH_ + hh)) * N_ + n) * HD_ + d] = (half_t)val;
          else
            ((half_t*)outp)[(((size_t)(bidx * NH_ + hh)) * HD_ + d) * N_ + n] = (half_t)val;
        }
      }
    }
  }
}

// ---------------------------------------------------------------------------
// RoPE, in-place on f16 q/k stored (b,h,n,d). Angle derived from the 2-D
// RoPE construction: j = h*HD + 2*d2; pos = (j<128 ? n/16 : n%16);
// theta = pos * 10000^-( (j&63)/64 ).
// ---------------------------------------------------------------------------
__global__ void rope_k(half_t* __restrict__ q, half_t* __restrict__ kk) {
  int idx = blockIdx.x * blockDim.x + threadIdx.x;
  if (idx >= B_ * NH_ * N_ * (HD_ / 2)) return;
  int d2 = idx & 15;
  int n  = (idx >> 4) & 255;
  int h  = (idx >> 12) & 7;
  int b  = idx >> 15;
  int j = h * HD_ + 2 * d2;
  float pos = (float)((j < 128) ? (n >> 4) : (n & 15));
  int t = j & 63;
  float ang = pos * powf(10000.f, -(float)t / 64.f);
  float cc = cosf(ang), ss = sinf(ang);
  size_t base = ((((size_t)b * NH_ + h) * N_) + n) * HD_ + 2 * d2;
  float x1 = (float)q[base], x2 = (float)q[base + 1];
  q[base]     = (half_t)(x1 * cc - x2 * ss);
  q[base + 1] = (half_t)(x1 * ss + x2 * cc);
  x1 = (float)kk[base]; x2 = (float)kk[base + 1];
  kk[base]     = (half_t)(x1 * cc - x2 * ss);
  kk[base + 1] = (half_t)(x1 * ss + x2 * cc);
}

// ---------------------------------------------------------------------------
// Attention: one wave per (b, h, 16-query tile). HD=32 => one WMMA k-step
// per 16x16 score tile. Scores live in LDS (never HBM); softmax uses
// shfl_xor(16) to combine the two half-row lanes; P@V accumulates from LDS
// A-fragments against transposed V.
// ---------------------------------------------------------------------------
__global__ __launch_bounds__(32)
void attention(const half_t* __restrict__ q, const half_t* __restrict__ k,
               const half_t* __restrict__ vT, half_t* __restrict__ o) {
  __shared__ __align__(32) float  sS[16 * 256];
  __shared__ __align__(32) half_t sP[16 * 256];
  const int lane = threadIdx.x & 31;
  const int hi = lane >> 4, l15 = lane & 15;
  const int qt = blockIdx.x & 15;
  const int h  = (blockIdx.x >> 4) & 7;
  const int b  = blockIdx.x >> 7;
  const float scale = 0.17677669529f;   // HD^-0.5

  const half_t* qr = q + ((((size_t)(b * NH_ + h)) * N_ + qt * 16 + l15)) * HD_ + hi * 16;
  union { v16h v; v8h h2[2]; } ua;
  ua.h2[0] = *(const v8h*)(qr);
  ua.h2[1] = *(const v8h*)(qr + 8);

  const half_t* kb = k + ((size_t)(b * NH_ + h)) * N_ * HD_;
  for (int kt = 0; kt < 16; ++kt) {
    v16h bv = *(const v16h*)(kb + (size_t)(kt * 16 + l15) * HD_ + hi * 16);
    v8f c = {};
    c = __builtin_amdgcn_wmma_f32_16x16x32_f16(false, ua.v, false, bv, (short)0, c, false, false);
#pragma unroll
    for (int r = 0; r < 8; ++r)
      sS[(hi * 8 + r) * 256 + kt * 16 + l15] = c[r] * scale;
  }
  __syncthreads();

  { // softmax: row = l15, each lane half covers 128 keys
    const int row = l15;
    float mx = -1e30f;
    for (int c0 = 0; c0 < 128; ++c0)
      mx = fmaxf(mx, sS[row * 256 + hi * 128 + c0]);
    mx = fmaxf(mx, __shfl_xor(mx, 16, 32));
    float s = 0.f;
    for (int c0 = 0; c0 < 128; ++c0) {
      float e = expf(sS[row * 256 + hi * 128 + c0] - mx);
      sS[row * 256 + hi * 128 + c0] = e;
      s += e;
    }
    s += __shfl_xor(s, 16, 32);
    const float inv = 1.f / s;
    for (int c0 = 0; c0 < 128; ++c0)
      sP[row * 256 + hi * 128 + c0] = (half_t)(sS[row * 256 + hi * 128 + c0] * inv);
  }
  __syncthreads();

  const half_t* vb = vT + ((size_t)(b * NH_ + h)) * HD_ * N_;
#pragma unroll
  for (int j = 0; j < 2; ++j) {
    const half_t* vr = vb + (size_t)(j * 16 + l15) * N_ + hi * 16;
    v8f c = {};
    for (int kc = 0; kc < 8; ++kc) {
      union { v16h v; v8h h2[2]; } pa;
      pa.h2[0] = *(const v8h*)&sP[l15 * 256 + kc * 32 + hi * 16];
      pa.h2[1] = *(const v8h*)&sP[l15 * 256 + kc * 32 + hi * 16 + 8];
      v16h bv = *(const v16h*)(vr + kc * 32);
      c = __builtin_amdgcn_wmma_f32_16x16x32_f16(false, pa.v, false, bv, (short)0, c, false, false);
    }
#pragma unroll
    for (int r = 0; r < 8; ++r) {
      const int qi = qt * 16 + hi * 8 + r;
      o[(((size_t)b * N_) + qi) * D_ + h * HD_ + j * 16 + l15] = (half_t)c[r];
    }
  }
}

// ---------------------------------------------------------------------------
extern "C" void kernel_launch(void* const* d_in, const int* in_sizes, int n_in,
                              void* d_out, int out_size, void* d_ws, size_t ws_size,
                              hipStream_t stream) {
  (void)in_sizes; (void)n_in; (void)out_size; (void)ws_size;
  const int*   g_grid  = (const int*)  d_in[0];
  const float* g_ce    = (const float*)d_in[1];
  const float* g_projw = (const float*)d_in[2];
  const float* g_projb = (const float*)d_in[3];
  const float* g_ln1g  = (const float*)d_in[4];
  const float* g_ln1b  = (const float*)d_in[5];
  const float* g_wq    = (const float*)d_in[6];
  const float* g_bq    = (const float*)d_in[7];
  const float* g_wk    = (const float*)d_in[8];
  const float* g_bk    = (const float*)d_in[9];
  const float* g_wv    = (const float*)d_in[10];
  const float* g_bv    = (const float*)d_in[11];
  const float* g_wo    = (const float*)d_in[12];
  const float* g_bo    = (const float*)d_in[13];
  const float* g_ln2g  = (const float*)d_in[14];
  const float* g_ln2b  = (const float*)d_in[15];
  const float* g_w1    = (const float*)d_in[16];
  const float* g_b1    = (const float*)d_in[17];
  const float* g_w2    = (const float*)d_in[18];
  const float* g_b2    = (const float*)d_in[19];
  const float* g_lnfg  = (const float*)d_in[20];
  const float* g_lnfb  = (const float*)d_in[21];

  char* wsp = (char*)d_ws;
  auto carve = [&](size_t bytes) -> char* {
    char* p = wsp; wsp += (bytes + 255) & ~(size_t)255; return p;
  };
  half_t* projWT = (half_t*)carve((size_t)1024 * 256 * 2);
  half_t* wqT = (half_t*)carve((size_t)L_ * D_ * D_ * 2);
  half_t* wkT = (half_t*)carve((size_t)L_ * D_ * D_ * 2);
  half_t* wvT = (half_t*)carve((size_t)L_ * D_ * D_ * 2);
  half_t* woT = (half_t*)carve((size_t)L_ * D_ * D_ * 2);
  half_t* w1T = (half_t*)carve((size_t)L_ * D_ * MLP_ * 2);
  half_t* w2T = (half_t*)carve((size_t)L_ * MLP_ * D_ * 2);
  half_t* ebuf  = (half_t*)carve((size_t)MT_ * 1024 * 2); // patch embed / MLP hidden
  half_t* xnbuf = (half_t*)carve((size_t)MT_ * D_ * 2);
  half_t* obuf  = (half_t*)carve((size_t)MT_ * D_ * 2);
  half_t* qbuf  = (half_t*)carve((size_t)MT_ * D_ * 2);
  half_t* kbuf  = (half_t*)carve((size_t)MT_ * D_ * 2);
  half_t* vTbuf = (half_t*)carve((size_t)MT_ * D_ * 2);
  float*  x     = (float*) carve((size_t)MT_ * D_ * 4);

  auto tr = [&](const float* W, half_t* WT, int K, int Nn) {
    int total = K * Nn;
    trans_k<<<(total + 255) / 256, 256, 0, stream>>>(W, WT, K, Nn);
  };
  tr(g_projw, projWT, 1024, 256);
  for (int i = 0; i < L_; ++i) {
    tr(g_wq + (size_t)i * D_ * D_,  wqT + (size_t)i * D_ * D_,  D_, D_);
    tr(g_wk + (size_t)i * D_ * D_,  wkT + (size_t)i * D_ * D_,  D_, D_);
    tr(g_wv + (size_t)i * D_ * D_,  wvT + (size_t)i * D_ * D_,  D_, D_);
    tr(g_wo + (size_t)i * D_ * D_,  woT + (size_t)i * D_ * D_,  D_, D_);
    tr(g_w1 + (size_t)i * D_ * MLP_, w1T + (size_t)i * D_ * MLP_, D_, MLP_);
    tr(g_w2 + (size_t)i * MLP_ * D_, w2T + (size_t)i * MLP_ * D_, MLP_, D_);
  }

  embed_k<<<(MT_ * 1024 + 255) / 256, 256, 0, stream>>>(g_grid, g_ce, ebuf);

  auto gblocks = [](int Mtot, int Nout) {
    int tiles = (Mtot / 32) * (Nout / 64);
    return (tiles + 7) / 8;
  };

  // x = patches @ proj_w + proj_b
  gemm_f16<0, false, false><<<gblocks(MT_, D_), 256, 0, stream>>>(
      ebuf, projWT, g_projb, nullptr, x, MT_, 1024, D_);

  for (int i = 0; i < L_; ++i) {
    ln_k<false><<<MT_ / 8, 256, 0, stream>>>(x, g_ln1g + i * D_, g_ln1b + i * D_, xnbuf, MT_);
    gemm_f16<2, false, false><<<gblocks(MT_, D_), 256, 0, stream>>>(
        xnbuf, wqT + (size_t)i * D_ * D_, g_bq + i * D_, nullptr, qbuf, MT_, D_, D_);
    gemm_f16<2, false, false><<<gblocks(MT_, D_), 256, 0, stream>>>(
        xnbuf, wkT + (size_t)i * D_ * D_, g_bk + i * D_, nullptr, kbuf, MT_, D_, D_);
    gemm_f16<3, false, false><<<gblocks(MT_, D_), 256, 0, stream>>>(
        xnbuf, wvT + (size_t)i * D_ * D_, g_bv + i * D_, nullptr, vTbuf, MT_, D_, D_);
    rope_k<<<(B_ * NH_ * N_ * (HD_ / 2) + 255) / 256, 256, 0, stream>>>(qbuf, kbuf);
    attention<<<B_ * NH_ * (N_ / 16), 32, 0, stream>>>(qbuf, kbuf, vTbuf, obuf);
    gemm_f16<0, false, true><<<gblocks(MT_, D_), 256, 0, stream>>>(
        obuf, woT + (size_t)i * D_ * D_, g_bo + i * D_, x, x, MT_, D_, D_);
    ln_k<false><<<MT_ / 8, 256, 0, stream>>>(x, g_ln2g + i * D_, g_ln2b + i * D_, xnbuf, MT_);
    gemm_f16<1, true, false><<<gblocks(MT_, MLP_), 256, 0, stream>>>(
        xnbuf, w1T + (size_t)i * D_ * MLP_, g_b1 + i * MLP_, nullptr, ebuf, MT_, D_, MLP_);
    gemm_f16<0, false, true><<<gblocks(MT_, D_), 256, 0, stream>>>(
        ebuf, w2T + (size_t)i * MLP_ * D_, g_b2 + i * D_, x, x, MT_, MLP_, D_);
  }

  ln_k<true><<<MT_ / 8, 256, 0, stream>>>(x, g_lnfg, g_lnfb, d_out, MT_);
}
